// AAttn_4707284156515
// MI455X (gfx1250) — matrix operand order repaired and checked by reading, not compiled, not run on hardware
//
#include <hip/hip_runtime.h>

// ---------- types ----------
typedef __bf16 bf16_t;
typedef __attribute__((ext_vector_type(16))) __bf16 v16bf;
typedef __attribute__((ext_vector_type(8)))  __bf16 v8bf;
typedef __attribute__((ext_vector_type(4)))  __bf16 v4bf;
typedef __attribute__((ext_vector_type(8)))  float   v8f;
typedef __attribute__((ext_vector_type(2)))  int     v2i;
typedef __attribute__((ext_vector_type(4)))  unsigned int v4u;
typedef __attribute__((ext_vector_type(8)))  int     v8i;
typedef __attribute__((ext_vector_type(4)))  int     v4i;

#define C_DIM 256
#define HW    4096
#define EPSF  1e-5f

// gfx1250 async memory->LDS path (ASYNCcnt), with safe fallback
#if __has_builtin(__builtin_amdgcn_global_load_async_to_lds_b64) && \
    __has_builtin(__builtin_amdgcn_s_wait_asynccnt)
#define HAVE_ASYNC_LDS 1
#else
#define HAVE_ASYNC_LDS 0
#endif

// gfx1250 Tensor Data Mover (TENSORcnt); arity differs by toolchain:
// ROCm 7.2 = 5 args; amdgpu-toolchain (ships the TDM header) = 6 args.
#if __has_builtin(__builtin_amdgcn_tensor_load_to_lds) && \
    __has_builtin(__builtin_amdgcn_s_wait_tensorcnt)
#define HAVE_TDM 1
#if __has_include(<hip/amd_detail/amd_gfx1250_TDM.h>)
#define TDM_SIX_ARGS 1
#else
#define TDM_SIX_ARGS 0
#endif
#else
#define HAVE_TDM 0
#endif

#if HAVE_ASYNC_LDS
// b64 async DMA: per-lane 8 bytes, global (AS1) -> LDS (AS3), tracked by ASYNCcnt
static __device__ inline void async_copy_b64(const void* g, void* l) {
  __builtin_amdgcn_global_load_async_to_lds_b64(
      (__attribute__((address_space(1))) v2i*)(uintptr_t)g,
      (__attribute__((address_space(3))) v2i*)(unsigned)(uintptr_t)l,  // low 32b = LDS offset
      0, 0);
}
#endif

#if HAVE_TDM
// 2-D TDM load: rows x width elements of 2-byte data, row stride in elements.
// D# packing per CDNA5 ISA 8.3/8.4 (group0 128b, group1 256b; groups 2/3 zero for 2-D).
static __device__ inline void tdm_load_2d(const void* gaddr, void* laddr,
                                          unsigned rows, unsigned width,
                                          unsigned stride_elts) {
  const unsigned long long ga = (unsigned long long)(uintptr_t)gaddr;
  const unsigned la = (unsigned)(uintptr_t)laddr;            // low 32b = LDS byte offset
  v4u g0;
  g0.x = 1u;                                                 // count=1 (valid user D#)
  g0.y = la;                                                 // lds_addr
  g0.z = (unsigned)(ga & 0xffffffffu);                       // global_addr[31:0]
  g0.w = (unsigned)((ga >> 32) & 0x01ffffffu) | (2u << 30);  // addr[56:32] | type=2
  v8i g1;
  g1[0] = (int)(1u << 16);                                   // data_size=1 (2 bytes)
  g1[1] = (int)(width << 16);                                // tensor_dim0[15:0]
  g1[2] = (int)(rows  << 16);                                // dim0 hi=0 | tensor_dim1[15:0]
  g1[3] = (int)(width << 16);                                // dim1 hi=0 | tile_dim0
  g1[4] = (int)rows;                                         // tile_dim1 (tile_dim2=0)
  g1[5] = (int)stride_elts;                                  // tensor_dim0_stride[31:0]
  g1[6] = 0;
  g1[7] = 0;
  v4i gz = {};
#if TDM_SIX_ARGS
  v8i gz8 = {};
  __builtin_amdgcn_tensor_load_to_lds(g0, g1, gz, gz, gz8, 0);
#else
  __builtin_amdgcn_tensor_load_to_lds(g0, g1, gz, gz, 0);
#endif
}
#endif

// ---------- WMMA fragment builders (contiguous k = 16*(lane/16) + s, wave32) ----------
static __device__ inline v16bf frag_a(const bf16_t* row, int lane) {
  const int base = (lane & 16) ? 16 : 0;
  union { v16bf v; v8bf h[2]; } u;
  u.h[0] = *(const v8bf*)(row + base);
  u.h[1] = *(const v8bf*)(row + base + 8);
  return u.v;
}
static __device__ inline v16bf frag_b_contig(const bf16_t* colrun, int lane) {
  const int off = (lane & 16) ? 16 : 0;
  union { v16bf v; v8bf h[2]; } u;
  u.h[0] = *(const v8bf*)(colrun + off);
  u.h[1] = *(const v8bf*)(colrun + off + 8);
  return u.v;
}
static __device__ inline v16bf frag_b_strided_f32(const float* p, int stride, int lane) {
  const int off = (lane & 16) ? 16 : 0;
  p += (size_t)off * stride;
  v16bf v;
#pragma unroll
  for (int i = 0; i < 16; ++i) v[i] = (__bf16)p[(size_t)i * stride];
  return v;
}
static __device__ inline v16bf frag_b_strided_bf16(const bf16_t* p, int stride, int lane) {
  const int off = (lane & 16) ? 16 : 0;
  p += (size_t)off * stride;
  v16bf v;
#pragma unroll
  for (int i = 0; i < 16; ++i) v[i] = p[(size_t)i * stride];
  return v;
}
static __device__ inline v8f wmma_bf16(v16bf a, v16bf b, v8f c) {
  return __builtin_amdgcn_wmma_f32_16x16x32_bf16(false, a, false, b, (short)0, c, false, false);
}
static __device__ inline float rmax16(float v) {
#pragma unroll
  for (int m = 8; m; m >>= 1) v = fmaxf(v, __shfl_xor(v, m, 32));
  return v;
}
static __device__ inline float rsum16(float v) {
#pragma unroll
  for (int m = 8; m; m >>= 1) v += __shfl_xor(v, m, 32);
  return v;
}

// ---------- kernel 1: qkv 1x1 conv (GEMM) + folded BN, scatter to q/k/v + vmap ----------
__global__ __launch_bounds__(256) void qkv_gemm_kernel(
    const float* __restrict__ x, const float* __restrict__ w,
    const float* __restrict__ gamma, const float* __restrict__ beta,
    const float* __restrict__ mean, const float* __restrict__ var,
    bf16_t* __restrict__ qb, bf16_t* __restrict__ kb, bf16_t* __restrict__ vb,
    float* __restrict__ vmap)
{
  __shared__ __align__(16) bf16_t wlds[16 * 256];
  const int tid = threadIdx.x;
  const int ocT = blockIdx.y;                       // 48 tiles of 16 output channels
  for (int i = tid; i < 16 * 256; i += 256)
    wlds[i] = (__bf16)w[(size_t)(ocT * 16 + (i >> 8)) * 256 + (i & 255)];
  __syncthreads();

  const int wave = tid >> 5, lane = tid & 31;
  const int col  = lane & 15;
  const int roff = (lane & 16) ? 8 : 0;
  const int pxT  = blockIdx.x * 8 + wave;           // 1024 pixel tiles of 16
  const int px0  = pxT * 16;
  const int b    = px0 >> 12;
  const int n0   = px0 & 4095;

  const float* xcol = x + (size_t)b * C_DIM * HW + n0 + col;
  v8f acc = {};
#pragma unroll
  for (int k0 = 0; k0 < 256; k0 += 32) {
    v16bf A  = frag_a(&wlds[col * 256 + k0], lane);
    v16bf Bf = frag_b_strided_f32(xcol + (size_t)k0 * HW, HW, lane);
    if (k0 + 32 < 256) __builtin_prefetch(xcol + (size_t)(k0 + 32) * HW, 0, 1);
    acc = wmma_bf16(A, Bf, acc);
  }

  const int n  = n0 + col;
  const int ba = b * 4 + (n >> 10);
  const int na = n & 1023;
#pragma unroll
  for (int gi = 0; gi < 8; ++gi) {
    const int oc  = ocT * 16 + gi + roff;
    const float inv = gamma[oc] * rsqrtf(var[oc] + EPSF);
    const float val = acc[gi] * inv + (beta[oc] - mean[oc] * inv);
    const int head = oc / 96, r = oc % 96;          // 3C split as (8 heads, 96)
    const size_t rowb = (((size_t)(ba * 8 + head)) * 1024 + na) * 32;
    if (r < 32)      qb[rowb + r]        = (__bf16)val;
    else if (r < 64) kb[rowb + (r - 32)] = (__bf16)val;
    else {
      vb[rowb + (r - 64)] = (__bf16)val;
      vmap[((size_t)b * C_DIM + head * 32 + (r - 64)) * HW + n] = val;
    }
  }
}

// ---------- kernel 2: fused flash attention, whole K/V panels resident in LDS ----------
// Dynamic LDS: K panel 64 KB + V panel 64 KB + P staging 8 KB = 136 KB (of 320 KB/WGP).
// K via one TENSOR_LOAD_TO_LDS (TDM, TENSORcnt); V via per-lane async b64 (ASYNCcnt).
#define KPAN 0
#define VPAN (32 * 1024)
#define PSTG (64 * 1024)
#define ATTN_LDS_BYTES ((64 * 1024 + 4096) * 2)

__global__ __launch_bounds__(256) void attn_kernel(
    const bf16_t* __restrict__ qb, const bf16_t* __restrict__ kb,
    const bf16_t* __restrict__ vb, float* __restrict__ og)
{
  extern __shared__ __align__(16) bf16_t smem[];
  bf16_t* klds = smem + KPAN;                         // [1024 keys][32 d]
  bf16_t* vlds = smem + VPAN;                         // [1024 keys][32 d]

  const int bh  = blockIdx.x;                          // ba*8 + h
  const int ba  = bh >> 3, h = bh & 7;
  const int tid = threadIdx.x, wave = tid >> 5, lane = tid & 31;
  const int col = lane & 15;
  const int roff = (lane & 16) ? 8 : 0;

  const bf16_t* qbase = qb + (size_t)bh * 1024 * 32;
  const bf16_t* kbase = kb + (size_t)bh * 1024 * 32;
  const bf16_t* vbase = vb + (size_t)bh * 1024 * 32;

  // ---- stage whole K panel via TDM (single descriptor-driven DMA) ----
#if HAVE_TDM
  if (wave == 0) {
    tdm_load_2d(kbase, klds, /*rows=*/1024, /*width=*/32, /*stride=*/32);
  }
#else
  for (int i = tid * 4; i < 32 * 1024; i += 1024) {
#if HAVE_ASYNC_LDS
    async_copy_b64(kbase + i, &klds[i]);
#else
    *(v4bf*)&klds[i] = *(const v4bf*)(kbase + i);
#endif
  }
#endif
  // ---- stage whole V panel via async global->LDS DMA ----
  for (int i = tid * 4; i < 32 * 1024; i += 1024) {
#if HAVE_ASYNC_LDS
    async_copy_b64(vbase + i, &vlds[i]);
#else
    *(v4bf*)&vlds[i] = *(const v4bf*)(vbase + i);
#endif
  }
#if HAVE_ASYNC_LDS
  __builtin_amdgcn_s_wait_asynccnt(0);
#endif
#if HAVE_TDM
  if (wave == 0) __builtin_amdgcn_s_wait_tensorcnt(0);
#endif
  __syncthreads();                                     // panels visible to all waves

  const int q0 = blockIdx.y * 128 + wave * 16;         // this wave's 16 queries
  const v16bf qA = frag_a(qbase + (size_t)(q0 + col) * 32, lane);

  v8f o0 = {}, o1 = {};
  const v8f zero = {};
  float mrow[8], lrow[8];
#pragma unroll
  for (int gi = 0; gi < 8; ++gi) { mrow[gi] = -1e30f; lrow[gi] = 0.f; }
  const float scale = 0.1767766952966369f;             // 32^-0.5
  bf16_t* pw = smem + PSTG + wave * (16 * 32);         // per-wave private P tile

  // barrier-free main loop: panels are read-only from here on
  for (int m0 = 0; m0 < 1024; m0 += 32) {
    v16bf kB0 = frag_b_contig(&klds[(m0 + col) * 32], lane);
    v16bf kB1 = frag_b_contig(&klds[(m0 + 16 + col) * 32], lane);
    v8f s0 = wmma_bf16(qA, kB0, zero);
    v8f s1 = wmma_bf16(qA, kB1, zero);

#pragma unroll
    for (int gi = 0; gi < 8; ++gi) {                    // online softmax per query row
      const float a0 = s0[gi] * scale, a1 = s1[gi] * scale;
      const float cmax = rmax16(fmaxf(a0, a1));
      const float mnew = fmaxf(mrow[gi], cmax);
      const float resc = __expf(mrow[gi] - mnew);
      mrow[gi] = mnew;
      const float e0 = __expf(a0 - mnew), e1 = __expf(a1 - mnew);
      lrow[gi] = lrow[gi] * resc + rsum16(e0 + e1);
      o0[gi] *= resc; o1[gi] *= resc;
      pw[(gi + roff) * 32 + col]      = (__bf16)e0;     // C-layout -> row-major LDS
      pw[(gi + roff) * 32 + 16 + col] = (__bf16)e1;
    }

    // AV: o[q, d] += P(16x32) x V^T(32x16); same-wave DS ops are in-order
    v16bf pA  = frag_a(pw + col * 32, lane);
    v16bf vB0 = frag_b_strided_bf16(&vlds[m0 * 32 + col], 32, lane);
    v16bf vB1 = frag_b_strided_bf16(&vlds[m0 * 32 + 16 + col], 32, lane);
    o0 = wmma_bf16(pA, vB0, o0);
    o1 = wmma_bf16(pA, vB1, o1);
  }

  const int b = ba >> 2;
  const int nbase = (ba & 3) * 1024 + q0;
#pragma unroll
  for (int gi = 0; gi < 8; ++gi) {
    const float invl = 1.f / lrow[gi];
    const int n = nbase + gi + roff;
    og[((size_t)b * C_DIM + h * 32 + col) * HW + n]      = o0[gi] * invl;
    og[((size_t)b * C_DIM + h * 32 + 16 + col) * HW + n] = o1[gi] * invl;
  }
}

// ---------- kernel 3: depthwise 7x7 PE conv + BN + residual add, pack bf16 ----------
__global__ __launch_bounds__(256) void pe_add_kernel(
    const float* __restrict__ vmap, const float* __restrict__ pw,
    const float* __restrict__ pg, const float* __restrict__ pb,
    const float* __restrict__ pm, const float* __restrict__ pv,
    const float* __restrict__ og, bf16_t* __restrict__ sb)
{
  const int idx = blockIdx.x * 256 + threadIdx.x;       // b*C*HW flat
  const int n = idx & 4095;
  const int c = (idx >> 12) & 255;
  const int b = idx >> 20;
  const int y = n >> 6, xq = n & 63;
  const float* wv  = pw + c * 49;
  const float* src = vmap + ((size_t)b * C_DIM + c) * HW;
  float acc = 0.f;
#pragma unroll
  for (int ky = 0; ky < 7; ++ky) {
    const int yy = y + ky - 3;
    if ((unsigned)yy < 64u) {
#pragma unroll
      for (int kx = 0; kx < 7; ++kx) {
        const int xx = xq + kx - 3;
        if ((unsigned)xx < 64u) acc += wv[ky * 7 + kx] * src[yy * 64 + xx];
      }
    }
  }
  const float inv = pg[c] * rsqrtf(pv[c] + EPSF);
  const float pe  = acc * inv + (pb[c] - pm[c] * inv);
  sb[idx] = (__bf16)(pe + og[idx]);
}

// ---------- kernel 4: proj 1x1 conv (GEMM) + BN -> f32 output ----------
__global__ __launch_bounds__(256) void proj_gemm_kernel(
    const bf16_t* __restrict__ sb, const float* __restrict__ w,
    const float* __restrict__ gamma, const float* __restrict__ beta,
    const float* __restrict__ mean, const float* __restrict__ var,
    float* __restrict__ out)
{
  __shared__ __align__(16) bf16_t wlds[16 * 256];
  const int tid = threadIdx.x;
  const int ocT = blockIdx.y;                           // 16 tiles
  for (int i = tid; i < 16 * 256; i += 256)
    wlds[i] = (__bf16)w[(size_t)(ocT * 16 + (i >> 8)) * 256 + (i & 255)];
  __syncthreads();

  const int wave = tid >> 5, lane = tid & 31;
  const int col  = lane & 15;
  const int roff = (lane & 16) ? 8 : 0;
  const int pxT  = blockIdx.x * 8 + wave;
  const int px0  = pxT * 16;
  const int b    = px0 >> 12;
  const int n0   = px0 & 4095;

  const bf16_t* scol = sb + (size_t)b * C_DIM * HW + n0 + col;
  v8f acc = {};
#pragma unroll
  for (int k0 = 0; k0 < 256; k0 += 32) {
    v16bf A  = frag_a(&wlds[col * 256 + k0], lane);
    v16bf Bf = frag_b_strided_bf16(scol + (size_t)k0 * HW, HW, lane);
    if (k0 + 32 < 256) __builtin_prefetch(scol + (size_t)(k0 + 32) * HW, 0, 1);
    acc = wmma_bf16(A, Bf, acc);
  }
  const int n = n0 + col;
#pragma unroll
  for (int gi = 0; gi < 8; ++gi) {
    const int oc = ocT * 16 + gi + roff;
    const float inv = gamma[oc] * rsqrtf(var[oc] + EPSF);
    out[((size_t)b * C_DIM + oc) * HW + n] = acc[gi] * inv + (beta[oc] - mean[oc] * inv);
  }
}

// ---------- launch ----------
extern "C" void kernel_launch(void* const* d_in, const int* in_sizes, int n_in,
                              void* d_out, int out_size, void* d_ws, size_t ws_size,
                              hipStream_t stream)
{
  (void)in_sizes; (void)n_in; (void)out_size; (void)ws_size;
  const float* x      = (const float*)d_in[0];
  const float* qkv_w  = (const float*)d_in[1];
  const float* qkv_g  = (const float*)d_in[2];
  const float* qkv_b  = (const float*)d_in[3];
  const float* qkv_m  = (const float*)d_in[4];
  const float* qkv_v  = (const float*)d_in[5];
  const float* pe_w   = (const float*)d_in[6];
  const float* pe_g   = (const float*)d_in[7];
  const float* pe_b   = (const float*)d_in[8];
  const float* pe_m   = (const float*)d_in[9];
  const float* pe_v   = (const float*)d_in[10];
  const float* proj_w = (const float*)d_in[11];
  const float* proj_g = (const float*)d_in[12];
  const float* proj_b = (const float*)d_in[13];
  const float* proj_m = (const float*)d_in[14];
  const float* proj_v = (const float*)d_in[15];

  char* ws = (char*)d_ws;                                // 64 MiB used
  bf16_t* qbuf = (bf16_t*)(ws);                          //  8 MiB  [16][8][1024][32] bf16
  bf16_t* kbuf = (bf16_t*)(ws + (size_t)8  * (1 << 20)); //  8 MiB
  bf16_t* vbuf = (bf16_t*)(ws + (size_t)16 * (1 << 20)); //  8 MiB
  float*  vmap = (float*) (ws + (size_t)24 * (1 << 20)); // 16 MiB  [4][256][4096] f32
  float*  og   = (float*) (ws + (size_t)40 * (1 << 20)); // 16 MiB  attention out f32
  bf16_t* sbuf = (bf16_t*)(ws + (size_t)56 * (1 << 20)); //  8 MiB  (o+pe) bf16

  qkv_gemm_kernel<<<dim3(128, 48), 256, 0, stream>>>(
      x, qkv_w, qkv_g, qkv_b, qkv_m, qkv_v, qbuf, kbuf, vbuf, vmap);
  attn_kernel<<<dim3(128, 8), 256, ATTN_LDS_BYTES, stream>>>(qbuf, kbuf, vbuf, og);
  pe_add_kernel<<<dim3(4 * 256 * 4096 / 256), 256, 0, stream>>>(
      vmap, pe_w, pe_g, pe_b, pe_m, pe_v, og, sbuf);
  proj_gemm_kernel<<<dim3(128, 16), 256, 0, stream>>>(
      sbuf, proj_w, proj_g, proj_b, proj_m, proj_v, (float*)d_out);
}
